// MambaLikePSI_49950469653081
// MI455X (gfx1250) — matrix operand order repaired and checked by reading, not compiled, not run on hardware
//
#include <hip/hip_runtime.h>

// ---------------------------------------------------------------------------
// MambaLikePSI for MI455X (gfx1250): f16 WMMA GEMMs (TDM-fed LDS tiles) +
// chunked parallel scan.
// ---------------------------------------------------------------------------

#define DIM   1024
#define STATE 1024
#define BATCH 4
#define SEQ   4096
#define ROWS  (BATCH * SEQ)        // 16384
#define NCAT  (DIM + DIM + DIM/4)  // 2304 = [W_B | W_C | W_dt1]
#define CHUNK 256
#define NCHUNK (SEQ / CHUNK)       // 16

typedef _Float16 v16h  __attribute__((ext_vector_type(16)));
typedef float    v8f   __attribute__((ext_vector_type(8)));
typedef unsigned int u32x4 __attribute__((ext_vector_type(4)));
typedef int      i32x4 __attribute__((ext_vector_type(4)));
typedef int      i32x8 __attribute__((ext_vector_type(8)));

union Frag16 { u32x4 q[2]; v16h h; };

// Tensor Data Mover availability (device pass only; host pass never codegens).
#if defined(__HIP_DEVICE_COMPILE__) && __has_builtin(__builtin_amdgcn_tensor_load_to_lds)
#define TDM_AVAIL 1
#else
#define TDM_AVAIL 0
#endif

// --------------------------- convert / pack kernels ------------------------

__global__ __launch_bounds__(256) void k_f32_to_f16(const float* __restrict__ in,
                                                    _Float16* __restrict__ out, size_t n) {
  size_t i = (size_t)blockIdx.x * blockDim.x + threadIdx.x;
  if (i < n) out[i] = (_Float16)in[i];
}

// WcatT[n][k] (n-major, k contiguous), n in [0,2304), k in [0,1024)
__global__ __launch_bounds__(256) void k_build_wcatT(const float* __restrict__ Wb,
                                                     const float* __restrict__ Wc,
                                                     const float* __restrict__ Wd1,
                                                     _Float16* __restrict__ WcatT) {
  int g = blockIdx.x * 256 + threadIdx.x;            // 2304*1024
  int k = g & (DIM - 1);
  int n = g >> 10;
  float v;
  if (n < DIM)            v = Wb[(size_t)k * STATE + n];
  else if (n < 2 * DIM)   v = Wc[(size_t)k * STATE + (n - DIM)];
  else                    v = Wd1[(size_t)k * (DIM / 4) + (n - 2 * DIM)];
  WcatT[(size_t)n * DIM + k] = (_Float16)v;
}

__global__ __launch_bounds__(256) void k_build_bcat(const float* __restrict__ bB,
                                                    const float* __restrict__ bC,
                                                    const float* __restrict__ bd1,
                                                    float* __restrict__ bcat) {
  int g = blockIdx.x * 256 + threadIdx.x;
  if (g >= NCAT) return;
  bcat[g] = (g < DIM) ? bB[g] : (g < 2 * DIM) ? bC[g - DIM] : bd1[g - 2 * DIM];
}

// W_dt2 [256,1024] -> Wd2T[n][k], n in [0,1024), k in [0,256)
__global__ __launch_bounds__(256) void k_build_wd2T(const float* __restrict__ W,
                                                    _Float16* __restrict__ out) {
  int g = blockIdx.x * 256 + threadIdx.x;            // 1024*256
  int k = g & 255;
  int n = g >> 8;
  out[(size_t)n * 256 + k] = (_Float16)W[(size_t)k * STATE + n];
}

// W_out [1024,1024] -> WoutT[n][k]
__global__ __launch_bounds__(256) void k_build_woutT(const float* __restrict__ W,
                                                     _Float16* __restrict__ out) {
  int g = blockIdx.x * 256 + threadIdx.x;            // 1024*1024
  int k = g & (DIM - 1);
  int n = g >> 10;
  out[(size_t)n * DIM + k] = (_Float16)W[(size_t)k * DIM + n];
}

// ------------------------------ TDM helpers --------------------------------
// Builds a 2D tile descriptor (D#) per CDNA5 ISA and
// issues TENSOR_LOAD_TO_LDS. Tile: 32 f16 along K x 128 rows, row stride = K.
// LDS padding: pad_interval=3 (every 16 DWORDs = 64B row) + pad_amount=3
// (4 DWORDs = 16B) reproduces the 40-half (80B) padded LDS row stride.

#if TDM_AVAIL
__device__ __forceinline__ unsigned int tdm_sgpr(unsigned int v) {
  return __builtin_amdgcn_readfirstlane(v);
}

__device__ __forceinline__ void tdm_load_tile_f16(unsigned int ldsAddr,
                                                  const _Float16* gptr,
                                                  unsigned int tensorK,
                                                  unsigned int rows) {
  unsigned long long ga = (unsigned long long)(uintptr_t)gptr;
  u32x4 g0;
  g0.x = tdm_sgpr(1u);                                       // count=1, user mode
  g0.y = tdm_sgpr(ldsAddr);                                  // lds_addr [63:32]
  g0.z = tdm_sgpr((unsigned int)(ga & 0xFFFFFFFFu));         // global_addr lo
  g0.w = tdm_sgpr((unsigned int)((ga >> 32) & 0x1FFFFFFu)    // global_addr hi
                  | (2u << 30));                             // type=2 ("image")
  // group1: data_size=2B, pad_enable, pad_interval=16dw, pad_amount=4dw
  unsigned int w0 = (1u << 16) | (1u << 20) | (3u << 22) | (3u << 25);
  unsigned int td0 = tensorK;                                // tensor_dim0
  unsigned int td1 = rows;                                   // tensor_dim1
  unsigned int w1 = (td0 & 0xFFFFu) << 16;                   // dim0 bits 79:48
  unsigned int w2 = (td0 >> 16) | ((td1 & 0xFFFFu) << 16);   // dim1 bits 111:80
  unsigned int w3 = (td1 >> 16) | (32u << 16);               // tile_dim0=32
  unsigned int w4 = rows & 0xFFFFu;                          // tile_dim1, tile_dim2=0
  unsigned long long stride = tensorK;                       // dim0 stride (elems)
  unsigned int w5 = (unsigned int)(stride & 0xFFFFFFFFull);
  unsigned int w6 = (unsigned int)((stride >> 32) & 0xFFFFull); // dim1_stride=0
  unsigned int w7 = 0;
  i32x8 g1;
  g1[0] = (int)tdm_sgpr(w0); g1[1] = (int)tdm_sgpr(w1);
  g1[2] = (int)tdm_sgpr(w2); g1[3] = (int)tdm_sgpr(w3);
  g1[4] = (int)tdm_sgpr(w4); g1[5] = (int)tdm_sgpr(w5);
  g1[6] = (int)tdm_sgpr(w6); g1[7] = (int)tdm_sgpr(w7);
  i32x4 z4 = {0, 0, 0, 0};
#if defined(__clang_major__) && (__clang_major__ >= 23)
  i32x8 z8 = {0, 0, 0, 0, 0, 0, 0, 0};
  __builtin_amdgcn_tensor_load_to_lds(g0, g1, z4, z4, z8, 0);
#else
  __builtin_amdgcn_tensor_load_to_lds(g0, g1, z4, z4, 0);
#endif
}
#endif // TDM_AVAIL

// ------------------------------ WMMA GEMM ----------------------------------
// C[tileM:+128, tileN:+128] = A[M,K] (row-major f16) x Bt[N,K] (n-major f16)
// 8 waves/block; wave w owns rows [w*16, w*16+16) x all 128 cols.
// LDS tiles padded to 40 halves/row (80B): conflict-free, 16B aligned frags.

#define LDS_STRIDE 40

template <int EPI>
__global__ __launch_bounds__(256)
void gemm_wmma(const _Float16* __restrict__ A, const _Float16* __restrict__ Bt,
               int K,
               const float* __restrict__ bias,   // per-col bias
               const float* __restrict__ aux,    // EPI1/3: x(f32); EPI2: omega_base
               const float* __restrict__ aux2,   // EPI3: D
               float* __restrict__ out0,         // EPI1: content; EPI2: omega; EPI3: y
               float* __restrict__ out1,         // EPI1: Cm
               _Float16* __restrict__ out2)      // EPI1: t1 (f16)
{
  __shared__ _Float16 As[128 * LDS_STRIDE];
  __shared__ _Float16 Bs[128 * LDS_STRIDE];

  const int tileM = blockIdx.y * 128;
  const int tileN = blockIdx.x * 128;
  const int lane  = threadIdx.x & 31;
  const int wave  = threadIdx.x >> 5;
  const int h     = lane >> 4;        // half-wave select

  v8f acc[8] = {};

#if TDM_AVAIL
  const unsigned int asBase = (unsigned int)(uintptr_t)(&As[0]);
  const unsigned int bsBase = (unsigned int)(uintptr_t)(&Bs[0]);
#endif

  for (int k0 = 0; k0 < K; k0 += 32) {
    __syncthreads();
#if TDM_AVAIL
    // One DMA descriptor per tile, issued by wave 0 only (scalar-skipped
    // branch for other waves); hardware pads rows to the 80B LDS stride.
    if (wave == 0) {
      tdm_load_tile_f16(asBase, A + (size_t)tileM * K + k0, (unsigned)K, 128u);
      tdm_load_tile_f16(bsBase, Bt + (size_t)tileN * K + k0, (unsigned)K, 128u);
      __builtin_amdgcn_s_wait_tensorcnt((short)0);
    }
#else
    // Fallback: cooperative tile load, 512 u32x4 per tile.
    for (int i = threadIdx.x; i < 512; i += 256) {
      int row = i >> 2;
      int kc  = (i & 3) << 3;
      *(u32x4*)&As[row * LDS_STRIDE + kc] =
          *(const u32x4*)&A[(size_t)(tileM + row) * K + k0 + kc];
      *(u32x4*)&Bs[row * LDS_STRIDE + kc] =
          *(const u32x4*)&Bt[(size_t)(tileN + row) * K + k0 + kc];
    }
#endif
    __syncthreads();

    // A fragment (16x32, 16-bit): lanes 0-15 hold K [0..8)+[16..24), lanes
    // 16-31 hold K [8..16)+[24..32) for row = wave*16 + lane%16.
    Frag16 fa;
    int mrow = wave * 16 + (lane & 15);
    fa.q[0] = *(const u32x4*)&As[mrow * LDS_STRIDE + 8 * h];
    fa.q[1] = *(const u32x4*)&As[mrow * LDS_STRIDE + 16 + 8 * h];

    // Preload all 8 B fragments so DS loads can overlap the WMMA burst.
    Frag16 fb[8];
#pragma unroll
    for (int nb = 0; nb < 8; ++nb) {
      int ncol = nb * 16 + (lane & 15);
      int koff = 8 * h;
      fb[nb].q[0] = *(const u32x4*)&Bs[ncol * LDS_STRIDE + koff];
      fb[nb].q[1] = *(const u32x4*)&Bs[ncol * LDS_STRIDE + koff + 16];
    }
#pragma unroll
    for (int nb = 0; nb < 8; ++nb) {
      acc[nb] = __builtin_amdgcn_wmma_f32_16x16x32_f16(
          false, fa.h, false, fb[nb].h, (short)0, acc[nb], false, false);
    }
  }

  // Epilogue. C/D layout: lane%16 = col, VGPR r -> row r + 8*(lane/16).
  const int rowBase = tileM + wave * 16 + (h << 3);
  const int colLane = lane & 15;
#pragma unroll
  for (int nb = 0; nb < 8; ++nb) {
    int col = tileN + nb * 16 + colLane;
#pragma unroll
    for (int r = 0; r < 8; ++r) {
      int row = rowBase + r;
      float v = acc[nb][r];
      if constexpr (EPI == 1) {
        float vb = v + bias[col];
        if (col < STATE) {
          out0[(size_t)row * STATE + col] = vb * aux[(size_t)row * DIM + col];
        } else if (col < 2 * STATE) {
          out1[(size_t)row * STATE + (col - STATE)] = vb;
        } else {
          out2[(size_t)row * (DIM / 4) + (col - 2 * STATE)] = (_Float16)vb;
        }
      } else if constexpr (EPI == 2) {
        float vb = v + bias[col];
        float sp = (vb > 20.0f) ? vb : log1pf(expf(vb));   // softplus
        out0[(size_t)row * STATE + col] = aux[col] * sp;   // omega
      } else {
        out0[(size_t)row * DIM + col] =
            v + bias[col] + aux2[col] * aux[(size_t)row * DIM + col];
      }
    }
  }
}

// ------------------------------ chunked scan -------------------------------
// grid over (b, chunk, ch): 4*16*1024 = 65536 threads for per-chunk passes.

__global__ __launch_bounds__(256) void scan_chunk_omega(const float* __restrict__ omega,
                                                        float* __restrict__ oc) {
  int g = blockIdx.x * 256 + threadIdx.x;
  int ch = g & (STATE - 1);
  int chunk = (g >> 10) & (NCHUNK - 1);
  int b = g >> 14;
  size_t base = ((size_t)b * SEQ + (size_t)chunk * CHUNK) * STATE + ch;
  float s = 0.0f;
  for (int i = 0; i < CHUNK; ++i) s += omega[base + (size_t)i * STATE];
  oc[((size_t)b * NCHUNK + chunk) * STATE + ch] = s;
}

__global__ __launch_bounds__(256) void scan_phibase(const float* __restrict__ oc,
                                                    float* __restrict__ pb) {
  int g = blockIdx.x * 256 + threadIdx.x;   // 4096
  int ch = g & (STATE - 1);
  int b = g >> 10;
  float run = 0.0f;
  for (int c = 0; c < NCHUNK; ++c) {
    size_t i = ((size_t)b * NCHUNK + c) * STATE + ch;
    pb[i] = run;
    run += oc[i];
  }
}

__global__ __launch_bounds__(256) void scan_uv(const float* __restrict__ omega,
                                               const float* __restrict__ content,
                                               const float* __restrict__ pb,
                                               float* __restrict__ Uc,
                                               float* __restrict__ Vc) {
  int g = blockIdx.x * 256 + threadIdx.x;
  int ch = g & (STATE - 1);
  int chunk = (g >> 10) & (NCHUNK - 1);
  int b = g >> 14;
  size_t ci = ((size_t)b * NCHUNK + chunk) * STATE + ch;
  size_t base = ((size_t)b * SEQ + (size_t)chunk * CHUNK) * STATE + ch;
  float phi = pb[ci], U = 0.0f, V = 0.0f;
  for (int i = 0; i < CHUNK; ++i) {
    size_t idx = base + (size_t)i * STATE;
    phi += omega[idx];
    float sn, cs;
    __sincosf(phi, &sn, &cs);
    float co = content[idx];
    U += co * cs;
    V += co * sn;
  }
  Uc[ci] = U;
  Vc[ci] = V;
}

__global__ __launch_bounds__(256) void scan_mbase(const float* __restrict__ Uc,
                                                  const float* __restrict__ Vc,
                                                  float* __restrict__ mrB,
                                                  float* __restrict__ miB) {
  int g = blockIdx.x * 256 + threadIdx.x;   // 4096
  int ch = g & (STATE - 1);
  int b = g >> 10;
  float ru = 0.0f, rv = 0.0f;
  for (int c = 0; c < NCHUNK; ++c) {
    size_t i = ((size_t)b * NCHUNK + c) * STATE + ch;
    mrB[i] = ru; miB[i] = rv;
    ru += Uc[i]; rv += Vc[i];
  }
}

__global__ __launch_bounds__(256) void scan_retrieve(const float* __restrict__ omega,
                                                     const float* __restrict__ content,
                                                     const float* __restrict__ Cm,
                                                     const float* __restrict__ pb,
                                                     const float* __restrict__ mrB,
                                                     const float* __restrict__ miB,
                                                     _Float16* __restrict__ reth) {
  int g = blockIdx.x * 256 + threadIdx.x;
  int ch = g & (STATE - 1);
  int chunk = (g >> 10) & (NCHUNK - 1);
  int b = g >> 14;
  size_t ci = ((size_t)b * NCHUNK + chunk) * STATE + ch;
  size_t base = ((size_t)b * SEQ + (size_t)chunk * CHUNK) * STATE + ch;
  float phi = pb[ci], mr = mrB[ci], mi = miB[ci];
  for (int i = 0; i < CHUNK; ++i) {
    size_t idx = base + (size_t)i * STATE;
    phi += omega[idx];
    float sn, cs;
    __sincosf(phi, &sn, &cs);
    float co = content[idx];
    mr += co * cs;
    mi += co * sn;
    float r = Cm[idx] * (mr * cs + mi * sn);
    reth[idx] = (_Float16)r;
  }
}

// ------------------------------ launcher -----------------------------------

static inline size_t align256(size_t x) { return (x + 255) & ~(size_t)255; }

extern "C" void kernel_launch(void* const* d_in, const int* in_sizes, int n_in,
                              void* d_out, int out_size, void* d_ws, size_t ws_size,
                              hipStream_t stream) {
  const float* x          = (const float*)d_in[0];
  const float* W_B        = (const float*)d_in[1];
  const float* b_B        = (const float*)d_in[2];
  const float* W_C        = (const float*)d_in[3];
  const float* b_C        = (const float*)d_in[4];
  const float* W_dt1      = (const float*)d_in[5];
  const float* b_dt1      = (const float*)d_in[6];
  const float* W_dt2      = (const float*)d_in[7];
  const float* b_dt2      = (const float*)d_in[8];
  const float* omega_base = (const float*)d_in[9];
  const float* Dv         = (const float*)d_in[10];
  const float* W_out      = (const float*)d_in[11];
  const float* b_out      = (const float*)d_in[12];
  float* y = (float*)d_out;

  // Workspace bump allocator
  char* w = (char*)d_ws;
  size_t off = 0;
  auto alloc = [&](size_t bytes) -> void* {
    void* p = w + off;
    off = align256(off + bytes);
    return p;
  };
  _Float16* xh    = (_Float16*)alloc((size_t)ROWS * DIM * 2);
  _Float16* WcatT = (_Float16*)alloc((size_t)NCAT * DIM * 2);
  float*    bcat  = (float*)   alloc((size_t)NCAT * 4);
  _Float16* Wd2T  = (_Float16*)alloc((size_t)STATE * (DIM / 4) * 2);
  _Float16* WoutT = (_Float16*)alloc((size_t)DIM * STATE * 2);
  float*    content = (float*) alloc((size_t)ROWS * STATE * 4);
  float*    Cm      = (float*) alloc((size_t)ROWS * STATE * 4);
  _Float16* t1h     = (_Float16*)alloc((size_t)ROWS * (DIM / 4) * 2);
  float*    omega   = (float*) alloc((size_t)ROWS * STATE * 4);
  _Float16* reth    = (_Float16*)alloc((size_t)ROWS * STATE * 2);
  const size_t nChunkElems = (size_t)BATCH * NCHUNK * STATE;   // 65536
  float* oc  = (float*)alloc(nChunkElems * 4);
  float* pb  = (float*)alloc(nChunkElems * 4);
  float* Uc  = (float*)alloc(nChunkElems * 4);
  float* Vc  = (float*)alloc(nChunkElems * 4);
  float* mrB = (float*)alloc(nChunkElems * 4);
  float* miB = (float*)alloc(nChunkElems * 4);
  (void)ws_size; (void)n_in; (void)in_sizes; (void)out_size;

  // 1) Precision convert / weight packing (one-shot, tiny vs GEMMs)
  k_f32_to_f16<<<(ROWS * DIM) / 256, 256, 0, stream>>>(x, xh, (size_t)ROWS * DIM);
  k_build_wcatT<<<(NCAT * DIM) / 256, 256, 0, stream>>>(W_B, W_C, W_dt1, WcatT);
  k_build_bcat<<<(NCAT + 255) / 256, 256, 0, stream>>>(b_B, b_C, b_dt1, bcat);
  k_build_wd2T<<<(STATE * (DIM / 4)) / 256, 256, 0, stream>>>(W_dt2, Wd2T);
  k_build_woutT<<<(DIM * STATE) / 256, 256, 0, stream>>>(W_out, WoutT);

  // 2) Fused projection GEMM: [16384,1024] x [1024,2304] -> content | Cm | t1
  gemm_wmma<1><<<dim3(NCAT / 128, ROWS / 128), 256, 0, stream>>>(
      xh, WcatT, DIM, bcat, x, nullptr, content, Cm, t1h);

  // 3) dt GEMM + softplus + omega: [16384,256] x [256,1024]
  gemm_wmma<2><<<dim3(STATE / 128, ROWS / 128), 256, 0, stream>>>(
      t1h, Wd2T, DIM / 4, b_dt2, omega_base, nullptr, omega, nullptr, nullptr);

  // 4) Chunked parallel scan over SEQ (phase + complex memory cumsums)
  scan_chunk_omega<<<(BATCH * NCHUNK * STATE) / 256, 256, 0, stream>>>(omega, oc);
  scan_phibase<<<(BATCH * STATE) / 256, 256, 0, stream>>>(oc, pb);
  scan_uv<<<(BATCH * NCHUNK * STATE) / 256, 256, 0, stream>>>(omega, content, pb, Uc, Vc);
  scan_mbase<<<(BATCH * STATE) / 256, 256, 0, stream>>>(Uc, Vc, mrB, miB);
  scan_retrieve<<<(BATCH * NCHUNK * STATE) / 256, 256, 0, stream>>>(
      omega, content, Cm, pb, mrB, miB, reth);

  // 5) Output GEMM + residual: y = retrieved @ W_out + b_out + D*x
  gemm_wmma<3><<<dim3(DIM / 128, ROWS / 128), 256, 0, stream>>>(
      reth, WoutT, STATE, b_out, x, Dv, y, nullptr, nullptr);
}
